// TreeToTreeAttention_43078521979640
// MI455X (gfx1250) — matrix operand order repaired and checked by reading, not compiled, not run on hardware
//
#include <hip/hip_runtime.h>
#include <hip/hip_bf16.h>
#include <math.h>

// ---------------------------------------------------------------------------
// TreeToTreeAttention on MI455X (gfx1250): fused flash-attention pipeline,
// all GEMMs via v_wmma_f32_16x16x32_bf16 (f32 accumulate).
// ---------------------------------------------------------------------------

typedef __bf16 bf16;
typedef __attribute__((ext_vector_type(16))) __bf16 v16bf;
typedef __attribute__((ext_vector_type(8)))  __bf16 v8bf;
typedef __attribute__((ext_vector_type(8)))  float  v8f;

#define HDIM   256
#define NENC   8192
#define TDEC   4096
#define SPLIT  8
#define NSPLIT (NENC / SPLIT)   // 1024 encoder nodes per split
#define TTILES (TDEC / 16)      // 256 decoder row-tiles

__device__ __forceinline__ bf16  f2bf(float f) { return (bf16)f; }

// A fragment (16x32, MxK): lane m = lane&15, group g = lane>>4.
// elements 0..7  -> K = 8g + 0..7,  elements 8..15 -> K = 8g + 16..23.
__device__ __forceinline__ v16bf load_frag_a(const bf16* __restrict__ p, int ld, int lane) {
    int m = lane & 15, g = lane >> 4;
    const bf16* r = p + (size_t)m * ld + g * 8;
    union { v16bf v; v8bf h[2]; } u;
    u.h[0] = *reinterpret_cast<const v8bf*>(r);
    u.h[1] = *reinterpret_cast<const v8bf*>(r + 16);
    return u.v;
}

// B fragment (32x16, KxN) sourced from Bt stored N-major [N][K]:
// lane n = lane&15 selects output column, elements 0..7 -> K = 8g + 0..7, 8..15 -> K = 8g + 16..23.
__device__ __forceinline__ v16bf load_frag_b(const bf16* __restrict__ p, int ld, int lane) {
    int n = lane & 15, g = lane >> 4;
    const bf16* r = p + (size_t)n * ld + g * 8;
    union { v16bf v; v8bf h[2]; } u;
    u.h[0] = *reinterpret_cast<const v8bf*>(r);
    u.h[1] = *reinterpret_cast<const v8bf*>(r + 16);
    return u.v;
}

__device__ __forceinline__ v8f wmma_bf16(v16bf a, v16bf b, v8f c) {
    return __builtin_amdgcn_wmma_f32_16x16x32_bf16(false, a, false, b, (short)0, c, false, false);
}

// ---------------------------------------------------------------------------
// Elementwise f32 -> bf16 convert
// ---------------------------------------------------------------------------
__global__ void k_convert(const float* __restrict__ src, bf16* __restrict__ dst, int n) {
    int i = blockIdx.x * blockDim.x + threadIdx.x;
    if (i < n) dst[i] = f2bf(src[i]);
}

// ---------------------------------------------------------------------------
// Tiled transpose + convert: src f32 [R][C] -> dst bf16 [C][R]
// grid (C/32, R/32), block (32, 8)
// ---------------------------------------------------------------------------
__global__ void k_transpose(const float* __restrict__ src, bf16* __restrict__ dst,
                            int R, int C) {
    __shared__ float tile[32][33];
    int c0 = blockIdx.x * 32, r0 = blockIdx.y * 32;
    int x = threadIdx.x, y = threadIdx.y;
    #pragma unroll
    for (int i = y; i < 32; i += 8)
        tile[i][x] = src[(size_t)(r0 + i) * C + c0 + x];
    __syncthreads();
    #pragma unroll
    for (int i = y; i < 32; i += 8)
        dst[(size_t)(c0 + i) * R + r0 + x] = f2bf(tile[x][i]);
}

// ---------------------------------------------------------------------------
// Kb = bf16(A @ W + bias):  A bf16 [M][K] row-major, Bt bf16 [256][K] (W^T),
// out bf16 [M][256].  Block = 8 waves: 2 row-tiles x 4 col-tiles -> 32x256.
// ---------------------------------------------------------------------------
__global__ void __launch_bounds__(256)
k_gemm_bias_bf16(const bf16* __restrict__ A, const bf16* __restrict__ Bt,
                 const float* __restrict__ bias, bf16* __restrict__ out, int K) {
    int lane = threadIdx.x & 31;
    int wave = threadIdx.x >> 5;
    int m0 = blockIdx.x * 32 + (wave >> 2) * 16;
    int c0 = (wave & 3) * 64;

    v8f acc[4] = {{}, {}, {}, {}};
    for (int k = 0; k < K; k += 32) {
        v16bf a = load_frag_a(A + (size_t)m0 * K + k, K, lane);
        #pragma unroll
        for (int f = 0; f < 4; f++) {
            v16bf b = load_frag_b(Bt + (size_t)(c0 + 16 * f) * K + k, K, lane);
            acc[f] = wmma_bf16(a, b, acc[f]);
        }
    }
    int g = lane >> 4, nn = lane & 15;
    #pragma unroll
    for (int f = 0; f < 4; f++) {
        int col = c0 + 16 * f + nn;
        float bv = bias[col];
        #pragma unroll
        for (int j = 0; j < 8; j++)
            out[(size_t)(m0 + 8 * g + j) * HDIM + col] = f2bf(acc[f][j] + bv);
    }
}

// ---------------------------------------------------------------------------
// Flash-attention partial: each wave handles 16 decoder rows x one N-split.
//   S = Q Kb^T (16 x 32 per chunk), online softmax, ctx += P V.
//   Vt = annotations^T bf16 [HDIM][NENC].
// Wave mapping: the 8 waves of a block share the SAME split s and walk the
// same K/V chunks in lockstep -> each 32KB chunk hits WGP$ 8x.
// Outputs unnormalized ctx_part f32 [SPLIT][TDEC][HDIM] + per-row m/l.
// ---------------------------------------------------------------------------
__global__ void __launch_bounds__(256)
k_flash_partial(const bf16* __restrict__ Qb, const bf16* __restrict__ Kb,
                const bf16* __restrict__ Vt,
                float* __restrict__ ctx_part, float* __restrict__ m_part,
                float* __restrict__ l_part) {
    __shared__ bf16 ldsP[8][16 * 32];      // per-wave P-tile (C-layout -> A-layout)
    int lane = threadIdx.x & 31;
    int wave = threadIdx.x >> 5;
    int wid  = blockIdx.x * 8 + wave;
    int t0   = (wid & (TTILES - 1)) * 16;  // consecutive t-tiles within a block
    int s    = wid >> 8;                   // split: constant across the block
    int n_begin = s * NSPLIT;
    int n_end   = n_begin + NSPLIT;
    int g = lane >> 4, nn = lane & 15;
    bf16* pbuf = &ldsP[wave][0];

    // Q fragments for 16 rows, full K=256, held in registers (64 VGPRs)
    v16bf qf[8];
    #pragma unroll
    for (int k = 0; k < 8; k++)
        qf[k] = load_frag_a(Qb + (size_t)t0 * HDIM + k * 32, HDIM, lane);

    v8f acc[16];
    #pragma unroll
    for (int f = 0; f < 16; f++) acc[f] = (v8f){};
    float mrow[8], lrow[8];
    #pragma unroll
    for (int j = 0; j < 8; j++) { mrow[j] = -INFINITY; lrow[j] = 0.f; }

    for (int n0 = n_begin; n0 < n_end; n0 += 32) {
        // ---- one-chunk-ahead prefetch into WGP$ (global_prefetch_b8) ----
        if (n0 + 32 < n_end) {
            __builtin_prefetch(Kb + (size_t)(n0 + 32 + lane) * HDIM, 0, 3);
            #pragma unroll
            for (int i = 0; i < 8; i++)
                __builtin_prefetch(Vt + (size_t)(lane + 32 * i) * NENC + n0 + 32, 0, 3);
        }
        // ---- S = Q * K^T for a 16x32 logits chunk ----
        v8f s0 = (v8f){}, s1 = (v8f){};
        #pragma unroll
        for (int k = 0; k < 8; k++) {
            v16bf b0 = load_frag_b(Kb + (size_t)n0 * HDIM + k * 32, HDIM, lane);
            v16bf b1 = load_frag_b(Kb + (size_t)(n0 + 16) * HDIM + k * 32, HDIM, lane);
            s0 = wmma_bf16(qf[k], b0, s0);
            s1 = wmma_bf16(qf[k], b1, s1);
        }
        // ---- online softmax: row stats via 16-lane butterflies ----
        float cm[8];
        #pragma unroll
        for (int j = 0; j < 8; j++) cm[j] = fmaxf(s0[j], s1[j]);
        #pragma unroll
        for (int off = 1; off < 16; off <<= 1)
            #pragma unroll
            for (int j = 0; j < 8; j++)
                cm[j] = fmaxf(cm[j], __shfl_xor(cm[j], off, 32));
        float alpha[8];
        #pragma unroll
        for (int j = 0; j < 8; j++) {
            float mn = fmaxf(mrow[j], cm[j]);
            alpha[j] = __expf(mrow[j] - mn);
            mrow[j]  = mn;
        }
        float p0[8], p1[8], rs[8];
        #pragma unroll
        for (int j = 0; j < 8; j++) {
            p0[j] = __expf(s0[j] - mrow[j]);
            p1[j] = __expf(s1[j] - mrow[j]);
            rs[j] = p0[j] + p1[j];
        }
        #pragma unroll
        for (int off = 1; off < 16; off <<= 1)
            #pragma unroll
            for (int j = 0; j < 8; j++)
                rs[j] += __shfl_xor(rs[j], off, 32);
        #pragma unroll
        for (int j = 0; j < 8; j++) lrow[j] = lrow[j] * alpha[j] + rs[j];
        #pragma unroll
        for (int f = 0; f < 16; f++)
            #pragma unroll
            for (int j = 0; j < 8; j++) acc[f][j] *= alpha[j];

        // ---- transpose P (C-layout) -> A-layout through per-wave LDS ----
        #pragma unroll
        for (int j = 0; j < 8; j++) {
            pbuf[(8 * g + j) * 32 + nn]      = f2bf(p0[j]);
            pbuf[(8 * g + j) * 32 + nn + 16] = f2bf(p1[j]);
        }
        asm volatile("s_wait_dscnt 0" ::: "memory");
        v16bf pa = load_frag_a(pbuf, 32, lane);

        // ---- ctx += P * V  (V^T rows = H output cols, K = n chunk) ----
        #pragma unroll
        for (int f = 0; f < 16; f++) {
            v16bf vb = load_frag_b(Vt + (size_t)(16 * f) * NENC + n0, NENC, lane);
            acc[f] = wmma_bf16(pa, vb, acc[f]);
        }
    }

    // ---- write partials ----
    size_t base = (size_t)s * TDEC + t0;
    #pragma unroll
    for (int f = 0; f < 16; f++)
        #pragma unroll
        for (int j = 0; j < 8; j++)
            ctx_part[(base + 8 * g + j) * HDIM + f * 16 + nn] = acc[f][j];
    if (nn == 0) {
        #pragma unroll
        for (int j = 0; j < 8; j++) {
            m_part[base + 8 * g + j] = mrow[j];
            l_part[base + 8 * g + j] = lrow[j];
        }
    }
}

// ---------------------------------------------------------------------------
// Combine the SPLIT partial softmaxes -> normalized ctx bf16 [TDEC][HDIM]
// ---------------------------------------------------------------------------
__global__ void k_combine(const float* __restrict__ ctx_part,
                          const float* __restrict__ m_part,
                          const float* __restrict__ l_part,
                          bf16* __restrict__ Ctxb) {
    int idx = blockIdx.x * 256 + threadIdx.x;   // t*HDIM + h
    int t = idx >> 8, h = idx & 255;
    float M = -INFINITY;
    #pragma unroll
    for (int s = 0; s < SPLIT; s++) M = fmaxf(M, m_part[s * TDEC + t]);
    float L = 0.f, v = 0.f;
    #pragma unroll
    for (int s = 0; s < SPLIT; s++) {
        float w = __expf(m_part[s * TDEC + t] - M);
        L += l_part[s * TDEC + t] * w;
        v += ctx_part[((size_t)s * TDEC + t) * HDIM + h] * w;
    }
    Ctxb[idx] = f2bf(v / L);
}

// ---------------------------------------------------------------------------
// et = tanh(Q @ W1^T + ctx @ W2^T + b_pre), f32 out [TDEC][HDIM]
// ---------------------------------------------------------------------------
__global__ void __launch_bounds__(256)
k_gemm2_bias_tanh(const bf16* __restrict__ A1, const bf16* __restrict__ B1t,
                  const bf16* __restrict__ A2, const bf16* __restrict__ B2t,
                  const float* __restrict__ bias, float* __restrict__ out) {
    int lane = threadIdx.x & 31;
    int wave = threadIdx.x >> 5;
    int m0 = blockIdx.x * 32 + (wave >> 2) * 16;
    int c0 = (wave & 3) * 64;

    v8f acc[4] = {{}, {}, {}, {}};
    for (int k = 0; k < HDIM; k += 32) {
        v16bf a = load_frag_a(A1 + (size_t)m0 * HDIM + k, HDIM, lane);
        #pragma unroll
        for (int f = 0; f < 4; f++) {
            v16bf b = load_frag_b(B1t + (size_t)(c0 + 16 * f) * HDIM + k, HDIM, lane);
            acc[f] = wmma_bf16(a, b, acc[f]);
        }
    }
    for (int k = 0; k < HDIM; k += 32) {
        v16bf a = load_frag_a(A2 + (size_t)m0 * HDIM + k, HDIM, lane);
        #pragma unroll
        for (int f = 0; f < 4; f++) {
            v16bf b = load_frag_b(B2t + (size_t)(c0 + 16 * f) * HDIM + k, HDIM, lane);
            acc[f] = wmma_bf16(a, b, acc[f]);
        }
    }
    int g = lane >> 4, nn = lane & 15;
    #pragma unroll
    for (int f = 0; f < 4; f++) {
        int col = c0 + 16 * f + nn;
        float bv = bias[col];
        #pragma unroll
        for (int j = 0; j < 8; j++)
            out[(size_t)(m0 + 8 * g + j) * HDIM + col] = tanhf(acc[f][j] + bv);
    }
}

// ---------------------------------------------------------------------------
extern "C" void kernel_launch(void* const* d_in, const int* in_sizes, int n_in,
                              void* d_out, int out_size, void* d_ws, size_t ws_size,
                              hipStream_t stream) {
    const float* annotations = (const float*)d_in[0];   // [NENC][HDIM]
    const float* dec         = (const float*)d_in[1];   // [TDEC][HDIM]
    const float* W_ah        = (const float*)d_in[2];   // [HDIM][HDIM]
    const float* b_ah        = (const float*)d_in[3];   // [HDIM]
    const float* W_pre       = (const float*)d_in[4];   // [2*HDIM][HDIM]
    const float* b_pre       = (const float*)d_in[5];   // [HDIM]
    float* out = (float*)d_out;

    char* ws = (char*)d_ws;
    const size_t MB = 1u << 20;
    bf16*  Qb       = (bf16*)(ws + 0 * MB);        // 2 MB  [TDEC][HDIM]
    bf16*  Ab       = (bf16*)(ws + 2 * MB);        // 4 MB  [NENC][HDIM]
    bf16*  At       = (bf16*)(ws + 6 * MB);        // 4 MB  [HDIM][NENC]
    bf16*  Kb       = (bf16*)(ws + 10 * MB);       // 4 MB  [NENC][HDIM]
    bf16*  Waht     = (bf16*)(ws + 14 * MB);       // 128 KB
    bf16*  W1t      = (bf16*)(ws + 14 * MB + 256 * 256 * 2);
    bf16*  W2t      = (bf16*)(ws + 14 * MB + 2 * 256 * 256 * 2);
    bf16*  Ctxb     = (bf16*)(ws + 15 * MB);       // 2 MB  [TDEC][HDIM]
    float* ctx_part = (float*)(ws + 17 * MB);      // 32 MB [SPLIT][TDEC][HDIM]
    float* m_part   = (float*)(ws + 49 * MB);      // 128 KB
    float* l_part   = (float*)(ws + 49 * MB + SPLIT * TDEC * 4);

    dim3 tb(32, 8);
    // 1) precision conversions + transposes
    k_convert<<<(TDEC * HDIM) / 256, 256, 0, stream>>>(dec, Qb, TDEC * HDIM);
    k_convert<<<(NENC * HDIM) / 256, 256, 0, stream>>>(annotations, Ab, NENC * HDIM);
    k_transpose<<<dim3(HDIM / 32, NENC / 32), tb, 0, stream>>>(annotations, At, NENC, HDIM);
    k_transpose<<<dim3(HDIM / 32, HDIM / 32), tb, 0, stream>>>(W_ah, Waht, HDIM, HDIM);
    k_transpose<<<dim3(HDIM / 32, HDIM / 32), tb, 0, stream>>>(W_pre, W1t, HDIM, HDIM);
    k_transpose<<<dim3(HDIM / 32, HDIM / 32), tb, 0, stream>>>(W_pre + HDIM * HDIM, W2t, HDIM, HDIM);
    // 2) Kb = bf16(annotations @ W_ah + b_ah)
    k_gemm_bias_bf16<<<NENC / 32, 256, 0, stream>>>(Ab, Waht, b_ah, Kb, HDIM);
    // 3) flash attention partials (2048 waves; block-local K/V chunk sharing)
    k_flash_partial<<<TTILES * SPLIT / 8, 256, 0, stream>>>(Qb, Kb, At,
                                                            ctx_part, m_part, l_part);
    // 4) combine splits
    k_combine<<<(TDEC * HDIM) / 256, 256, 0, stream>>>(ctx_part, m_part, l_part, Ctxb);
    // 5) et = tanh(Q @ W1^T + ctx @ W2^T + b_pre)
    k_gemm2_bias_tanh<<<TDEC / 32, 256, 0, stream>>>(Qb, W1t, Ctxb, W2t, b_pre, out);
}
